// MultiHeadLatentAttention_4544075399611
// MI455X (gfx1250) — matrix operand order repaired and checked by reading, not compiled, not run on hardware
//
#include <hip/hip_runtime.h>
#include <hip/hip_bf16.h>

typedef __bf16 bf16_t;
typedef __attribute__((ext_vector_type(16))) __bf16 bf16x16;
typedef __attribute__((ext_vector_type(8)))  __bf16 bf16x8;
typedef __attribute__((ext_vector_type(8)))  float  f32x8;

#define S_LEN  2048
#define DMODEL 2048
#define NHEADS 16
#define HDIM   128
#define BATCH  2
#define LATENT 256

__device__ __forceinline__ bf16_t f2bf(float f) {
  unsigned u = __builtin_bit_cast(unsigned, f);
  u += 0x7FFFu + ((u >> 16) & 1u);            // round-to-nearest-even
  unsigned short s = (unsigned short)(u >> 16);
  return __builtin_bit_cast(bf16_t, s);
}

__device__ __forceinline__ bf16x16 ldA(const bf16_t* row, int k0, int kh) {
  bf16x8 lo = *(const bf16x8*)(row + k0 + kh * 16);
  bf16x8 hi = *(const bf16x8*)(row + k0 + kh * 16 + 8);
  return __builtin_shufflevector(lo, hi, 0,1,2,3,4,5,6,7,8,9,10,11,12,13,14,15);
}

// ---------------- elementwise fp32 -> bf16 ----------------
__global__ __launch_bounds__(256) void cvt_f32_bf16(const float* __restrict__ in,
                                                    bf16_t* __restrict__ out, long n) {
  long i = (long)blockIdx.x * blockDim.x + threadIdx.x;
  long stride = (long)gridDim.x * blockDim.x;
  for (; i < n; i += stride) out[i] = f2bf(in[i]);
}

// ---------------- tiled transpose fp32[R,C] -> bf16[C,R] ----------------
__global__ __launch_bounds__(256) void transpose_f32_bf16(const float* __restrict__ in,
                                                          bf16_t* __restrict__ out,
                                                          int R, int C) {
  __shared__ float tile[32][33];
  int c0 = blockIdx.x * 32, r0 = blockIdx.y * 32;
  for (int i = 0; i < 32; i += 8) {
    int r = r0 + threadIdx.y + i, c = c0 + threadIdx.x;
    if (r < R && c < C) tile[threadIdx.y + i][threadIdx.x] = in[(size_t)r * C + c];
  }
  __syncthreads();
  for (int i = 0; i < 32; i += 8) {
    int oc = r0 + threadIdx.x;        // out col = in row
    int orow = c0 + threadIdx.y + i;  // out row = in col
    if (orow < C && oc < R) out[(size_t)orow * R + oc] = f2bf(tile[threadIdx.x][threadIdx.y + i]);
  }
}

// ---------------- WMMA GEMM: C[M,N] = A[M,K] * BT[N,K]^T ----------------
// Each wave computes a 32x32 tile as 2x2 WMMA fragments (register blocking:
// 2 b128 loads per WMMA instead of 4).
// mode 0: store bf16 row-major [M,N]
// mode 1: store bf16 transposed-per-head vT[b,h,hd,s]   (M=B*S rows, N=DMODEL cols)
// mode 2: store fp32 row-major [M,N] + bias[n]
__global__ __launch_bounds__(256) void gemm_wmma_bf16(const bf16_t* __restrict__ A,
                                                      const bf16_t* __restrict__ BT,
                                                      void* __restrict__ C,
                                                      int M, int N, int K, int mode,
                                                      const float* __restrict__ bias) {
  int lane = threadIdx.x & 31;
  int w    = threadIdx.x >> 5;
  int ntn  = N >> 5;                       // 32-wide n-tiles
  int ntiles = (M >> 5) * ntn;
  int tile = blockIdx.x * 8 + w;
  if (tile >= ntiles) return;
  int tm = tile / ntn, tn = tile - tm * ntn;
  int l16 = lane & 15, kh = lane >> 4;

  const bf16_t* arow0 = A  + (size_t)(tm * 32 + l16) * K;
  const bf16_t* arow1 = arow0 + (size_t)16 * K;
  const bf16_t* brow0 = BT + (size_t)(tn * 32 + l16) * K;
  const bf16_t* brow1 = brow0 + (size_t)16 * K;

  f32x8 acc[2][2] = {};
  for (int k0 = 0; k0 < K; k0 += 32) {
    bf16x16 a0 = ldA(arow0, k0, kh);
    bf16x16 a1 = ldA(arow1, k0, kh);
    bf16x16 b0 = *(const bf16x16*)(brow0 + k0 + kh * 16);
    bf16x16 b1 = *(const bf16x16*)(brow1 + k0 + kh * 16);
    acc[0][0] = __builtin_amdgcn_wmma_f32_16x16x32_bf16(false, a0, false, b0, (short)0, acc[0][0], false, false);
    acc[0][1] = __builtin_amdgcn_wmma_f32_16x16x32_bf16(false, a0, false, b1, (short)0, acc[0][1], false, false);
    acc[1][0] = __builtin_amdgcn_wmma_f32_16x16x32_bf16(false, a1, false, b0, (short)0, acc[1][0], false, false);
    acc[1][1] = __builtin_amdgcn_wmma_f32_16x16x32_bf16(false, a1, false, b1, (short)0, acc[1][1], false, false);
  }

#pragma unroll
  for (int mi = 0; mi < 2; ++mi) {
#pragma unroll
    for (int ni = 0; ni < 2; ++ni) {
      int n = tn * 32 + ni * 16 + l16;
      if (mode == 0) {
        bf16_t* o = (bf16_t*)C;
#pragma unroll
        for (int r = 0; r < 8; ++r) {
          int m = tm * 32 + mi * 16 + r + 8 * kh;
          o[(size_t)m * N + n] = f2bf(acc[mi][ni][r]);
        }
      } else if (mode == 1) {
        bf16_t* o = (bf16_t*)C;
        int h = n >> 7, hd = n & 127;
#pragma unroll
        for (int r = 0; r < 8; ++r) {
          int m = tm * 32 + mi * 16 + r + 8 * kh;
          int b_ = m >> 11, s = m & 2047;  // M = BATCH*S_LEN, S_LEN = 2048
          o[((size_t)(b_ * NHEADS + h) * HDIM + hd) * S_LEN + s] = f2bf(acc[mi][ni][r]);
        }
      } else {
        float* o = (float*)C;
        float bz = bias[n];
#pragma unroll
        for (int r = 0; r < 8; ++r) {
          int m = tm * 32 + mi * 16 + r + 8 * kh;
          o[(size_t)m * N + n] = acc[mi][ni][r] + bz;
        }
      }
    }
  }
}

// ---------------- flash-attention (causal) per 16-query tile per head ----------------
// grid = 512 blocks * 256 threads; all 8 waves of a block share qt (uniform barriers).
__global__ __launch_bounds__(256) void mla_flash_attn(const bf16_t* __restrict__ q,
                                                      const bf16_t* __restrict__ k,
                                                      const bf16_t* __restrict__ vT,
                                                      bf16_t* __restrict__ ctx) {
  __shared__ __align__(16) unsigned char plds[8][16 * 80];  // per-wave P staging (row stride 80B)
  int lane = threadIdx.x & 31;
  int w    = threadIdx.x >> 5;
  int qt   = blockIdx.x & 127;                 // 16-query tile index
  int bh   = ((blockIdx.x >> 7) << 3) + w;     // 0..31
  int b    = bh >> 4, h = bh & 15;
  int l16  = lane & 15, kh = lane >> 4;
  const float sc  = 0.088388347648318447f;     // 1/sqrt(HDIM)
  const float NEG = -__builtin_inff();

  // q A-fragments: 4 chunks of K=32 over head_dim
  bf16x16 qa[4];
  {
    const bf16_t* qrow = q + ((size_t)(b * S_LEN + qt * 16 + l16) * DMODEL + h * HDIM);
#pragma unroll
    for (int c = 0; c < 4; ++c) qa[c] = ldA(qrow, c * 32, kh);
  }

  f32x8 acc[8] = {};                // ctx accumulator: 8 hd-tiles of 16 cols
  float mrow[8], lrow[8];
#pragma unroll
  for (int r = 0; r < 8; ++r) { mrow[r] = NEG; lrow[r] = 0.f; }

  int nkt = (qt >> 1) + 1;          // 32-key blocks covering causal range
  for (int kt = 0; kt < nkt; ++kt) {
    int kbase = kt * 32;
    // ---- scores: two 16x16 tiles, K = HDIM via 4 WMMAs each ----
    f32x8 s0 = {}, s1 = {};
    const bf16_t* kr0 = k + ((size_t)(b * S_LEN + kbase + l16) * DMODEL + h * HDIM + kh * 16);
    const bf16_t* kr1 = kr0 + (size_t)16 * DMODEL;
#pragma unroll
    for (int c = 0; c < 4; ++c) {
      bf16x16 kb0 = *(const bf16x16*)(kr0 + c * 32);
      s0 = __builtin_amdgcn_wmma_f32_16x16x32_bf16(false, qa[c], false, kb0, (short)0, s0, false, false);
      bf16x16 kb1 = *(const bf16x16*)(kr1 + c * 32);
      s1 = __builtin_amdgcn_wmma_f32_16x16x32_bf16(false, qa[c], false, kb1, (short)0, s1, false, false);
    }
    // ---- scale + causal mask + online softmax ----
    int kp0 = kbase + l16, kp1 = kbase + 16 + l16;
    float rmax[8];
#pragma unroll
    for (int r = 0; r < 8; ++r) {
      int qp = qt * 16 + r + 8 * kh;
      float v0 = (kp0 <= qp) ? s0[r] * sc : NEG;
      float v1 = (kp1 <= qp) ? s1[r] * sc : NEG;
      s0[r] = v0; s1[r] = v1;
      rmax[r] = fmaxf(v0, v1);
    }
#pragma unroll
    for (int off = 1; off < 16; off <<= 1)
#pragma unroll
      for (int r = 0; r < 8; ++r) rmax[r] = fmaxf(rmax[r], __shfl_xor(rmax[r], off, 32));

    float alpha[8], rsum[8];
#pragma unroll
    for (int r = 0; r < 8; ++r) {
      float mn = fmaxf(mrow[r], rmax[r]);
      alpha[r] = __expf(mrow[r] - mn);
      mrow[r] = mn;
      float p0 = __expf(s0[r] - mn);
      float p1 = __expf(s1[r] - mn);
      s0[r] = p0; s1[r] = p1;
      rsum[r] = p0 + p1;
    }
#pragma unroll
    for (int off = 1; off < 16; off <<= 1)
#pragma unroll
      for (int r = 0; r < 8; ++r) rsum[r] += __shfl_xor(rsum[r], off, 32);
#pragma unroll
    for (int r = 0; r < 8; ++r) lrow[r] = lrow[r] * alpha[r] + rsum[r];
#pragma unroll
    for (int t = 0; t < 8; ++t)
#pragma unroll
      for (int r = 0; r < 8; ++r) acc[t][r] *= alpha[r];

    // ---- transpose P (C-layout -> A-layout) via per-wave LDS ----
    __syncthreads();
    {
      unsigned char* base = plds[w];
#pragma unroll
      for (int r = 0; r < 8; ++r) {
        int m = r + 8 * kh;
        *(bf16_t*)(base + m * 80 + l16 * 2)      = f2bf(s0[r]);
        *(bf16_t*)(base + m * 80 + 32 + l16 * 2) = f2bf(s1[r]);
      }
    }
    __syncthreads();
    bf16x16 pa;
    {
      const unsigned char* ar = plds[w] + l16 * 80;
    bf16x8 plo = *(const bf16x8*)(ar + kh * 32);
    bf16x8 phi = *(const bf16x8*)(ar + kh * 32 + 16);
      pa = __builtin_shufflevector(plo, phi, 0,1,2,3,4,5,6,7,8,9,10,11,12,13,14,15);
    }
    // ---- acc += P(16x32) @ V(32x16) per hd-tile; vT rows are contiguous in s ----
    const bf16_t* vbase = vT + (size_t)(b * NHEADS + h) * HDIM * S_LEN + kbase + kh * 16;
#pragma unroll
    for (int t = 0; t < 8; ++t) {
      bf16x16 vb = *(const bf16x16*)(vbase + (size_t)(t * 16 + l16) * S_LEN);
      acc[t] = __builtin_amdgcn_wmma_f32_16x16x32_bf16(false, pa, false, vb, (short)0, acc[t], false, false);
    }
  }

  // ---- normalize and store ctx (bf16 row-major, heads re-interleaved) ----
  bf16_t* obase = ctx + (size_t)(b * S_LEN) * DMODEL + h * HDIM;
#pragma unroll
  for (int r = 0; r < 8; ++r) {
    float inv = 1.0f / lrow[r];
    int m = r + 8 * kh;
    int qpos = qt * 16 + m;
#pragma unroll
    for (int t = 0; t < 8; ++t)
      obase[(size_t)qpos * DMODEL + t * 16 + l16] = f2bf(acc[t][r] * inv);
  }
}

// ---------------- host orchestration ----------------
extern "C" void kernel_launch(void* const* d_in, const int* in_sizes, int n_in,
                              void* d_out, int out_size, void* d_ws, size_t ws_size,
                              hipStream_t stream) {
  const float* x    = (const float*)d_in[0];  // [2,2048,2048]
  const float* Wq   = (const float*)d_in[1];  // [2048,2048]
  const float* Wdkv = (const float*)d_in[2];  // [2048,256]
  const float* Wuk  = (const float*)d_in[3];  // [256,2048]
  const float* Wuv  = (const float*)d_in[4];  // [256,2048]
  const float* Wout = (const float*)d_in[5];  // [2048,2048]
  const float* bout = (const float*)d_in[6];  // [2048]
  float* out = (float*)d_out;                 // [2,2048,2048] fp32

  const int M = BATCH * S_LEN;                // 4096 flattened rows
  size_t off = 0;
  auto carve = [&](size_t elems) {
    bf16_t* p = (bf16_t*)((char*)d_ws + off);
    off += elems * sizeof(bf16_t);
    return p;
  };
  bf16_t* xb    = carve((size_t)M * DMODEL);        // x bf16
  bf16_t* WqT   = carve((size_t)DMODEL * DMODEL);   // [N=2048,K=2048]
  bf16_t* WdkvT = carve((size_t)LATENT * DMODEL);   // [N=256, K=2048]
  bf16_t* WukT  = carve((size_t)DMODEL * LATENT);   // [N=2048,K=256]
  bf16_t* WuvT  = carve((size_t)DMODEL * LATENT);   // [N=2048,K=256]
  bf16_t* WoutT = carve((size_t)DMODEL * DMODEL);   // [N=2048,K=2048]
  bf16_t* qb    = carve((size_t)M * DMODEL);        // q  [b,s,h*128+hd]
  bf16_t* lat   = carve((size_t)M * LATENT);        // latent [b*s, 256]
  bf16_t* kb    = carve((size_t)M * DMODEL);        // k  [b,s,h*128+hd]
  bf16_t* vTb   = carve((size_t)M * DMODEL);        // vT [b,h,hd,s]
  bf16_t* ctxb  = carve((size_t)M * DMODEL);        // ctx bf16
  (void)ws_size; (void)in_sizes; (void)n_in; (void)out_size;

  dim3 tb(32, 8);
  // converts / transposes
  cvt_f32_bf16<<<8192, 256, 0, stream>>>(x, xb, (long)M * DMODEL);
  transpose_f32_bf16<<<dim3(64, 64), tb, 0, stream>>>(Wq,   WqT,   DMODEL, DMODEL);
  transpose_f32_bf16<<<dim3(8,  64), tb, 0, stream>>>(Wdkv, WdkvT, DMODEL, LATENT);
  transpose_f32_bf16<<<dim3(64,  8), tb, 0, stream>>>(Wuk,  WukT,  LATENT, DMODEL);
  transpose_f32_bf16<<<dim3(64,  8), tb, 0, stream>>>(Wuv,  WuvT,  LATENT, DMODEL);
  transpose_f32_bf16<<<dim3(64, 64), tb, 0, stream>>>(Wout, WoutT, DMODEL, DMODEL);

  // projections: 32x32 wave tiles, 8 waves/block
  // q:     (4096/32)*(2048/32) = 8192 tiles -> 1024 blocks
  // lat:   (4096/32)*( 256/32) = 1024 tiles ->  128 blocks
  gemm_wmma_bf16<<<1024, 256, 0, stream>>>(xb,  WqT,   qb,  M, DMODEL, DMODEL, 0, nullptr); // q
  gemm_wmma_bf16<<<128,  256, 0, stream>>>(xb,  WdkvT, lat, M, LATENT, DMODEL, 0, nullptr); // latent
  gemm_wmma_bf16<<<1024, 256, 0, stream>>>(lat, WukT,  kb,  M, DMODEL, LATENT, 0, nullptr); // k
  gemm_wmma_bf16<<<1024, 256, 0, stream>>>(lat, WuvT,  vTb, M, DMODEL, LATENT, 1, nullptr); // v (transposed store)

  // causal flash attention: 2*16*128 = 4096 waves = 512 blocks
  mla_flash_attn<<<512, 256, 0, stream>>>(qb, kb, vTb, ctxb);

  // output projection + bias -> fp32
  gemm_wmma_bf16<<<1024, 256, 0, stream>>>(ctxb, WoutT, out, M, DMODEL, DMODEL, 2, bout);
}
